// CrossModalAttention_86440511799417
// MI455X (gfx1250) — matrix operand order repaired and checked by reading, not compiled, not run on hardware
//
#include <hip/hip_runtime.h>

typedef __attribute__((ext_vector_type(16))) _Float16 v16h;
typedef __attribute__((ext_vector_type(8)))  _Float16 v8h;
typedef __attribute__((ext_vector_type(8)))  float    v8f;

#define NHEADS 8
#define KD     16
#define DV     32
#define NPIX   4096   // 64*64
#define CDIM   256
#define BATCH  2

__device__ __forceinline__ v8f wmma_f16(v16h a, v16h b, v8f c) {
  // D = A(16x32 f16) * B(32x16 f16) + C(16x16 f32)
  return __builtin_amdgcn_wmma_f32_16x16x32_f16(false, a, false, b, (short)0, c,
                                                false, false);
}

__device__ __forceinline__ v16h cat8(v8h lo, v8h hi) {
  return __builtin_shufflevector(lo, hi, 0, 1, 2, 3, 4, 5, 6, 7,
                                 8, 9, 10, 11, 12, 13, 14, 15);
}

// ---------------------------------------------------------------------------
// Prep: fold BN scale into conv weight, cast to f16.  Wo[o,c] = W[o,c]*s[o]
// ---------------------------------------------------------------------------
__global__ void k_fold_w(const float* __restrict__ W, const float* __restrict__ s,
                         _Float16* __restrict__ Wo, int O, int C) {
  int i = blockIdx.x * blockDim.x + threadIdx.x;
  if (i < O * C) {
    int o = i / C;
    Wo[i] = (_Float16)(W[i] * s[o]);
  }
}

// f32 [B][C][N]  ->  f16 [B][N][C]  (pixel-major so GEMM B-fragments are
// per-lane contiguous 32B loads)
__global__ void k_cvt_T(const float* __restrict__ x, _Float16* __restrict__ y) {
  int i = blockIdx.x * blockDim.x + threadIdx.x;       // output-linear index
  if (i < BATCH * NPIX * CDIM) {
    const int c   = i % CDIM;
    const int rem = i / CDIM;
    const int n   = rem % NPIX;
    const int b   = rem / NPIX;
    y[i] = (_Float16)x[((size_t)b * CDIM + c) * NPIX + n];
  }
}

// ---------------------------------------------------------------------------
// QKV projection via WMMA. One wave = 16(out-ch) x 16(pixel) f32 tile,
// K-depth 256 in 8 steps. t<4096 -> Q (rgb), <8192 -> K, else V (edge).
// Layouts:  Qh,Kh : [B*8][n][kd] f16      Vt : [B*8][d][n] f16
// ---------------------------------------------------------------------------
__global__ void k_qkv(const _Float16* __restrict__ xrT, const _Float16* __restrict__ xeT,
                      const _Float16* __restrict__ Wq2, const _Float16* __restrict__ Wk2,
                      const _Float16* __restrict__ Wv2,
                      const float* __restrict__ bq, const float* __restrict__ bk,
                      const float* __restrict__ bv,
                      _Float16* __restrict__ Qh, _Float16* __restrict__ Kh,
                      _Float16* __restrict__ Vt) {
  const int wv   = threadIdx.x >> 5;
  const int lane = threadIdx.x & 31;
  const int hi   = lane >> 4;
  const int lm   = lane & 15;
  const int t    = blockIdx.x * 8 + wv;        // wave-tile id, 0..16383

  int which, ot, pt;
  if (t < 8 * 512)       { which = 0; ot = t >> 9;              pt = t & 511; }
  else if (t < 16 * 512) { which = 1; int u = t - 8 * 512;  ot = u >> 9; pt = u & 511; }
  else                   { which = 2; int u = t - 16 * 512; ot = u >> 9; pt = u & 511; }

  const _Float16* X    = (which == 0) ? xrT : xeT;
  const _Float16* W    = (which == 0) ? Wq2 : (which == 1 ? Wk2 : Wv2);
  const float*    bias = (which == 0) ? bq  : (which == 1 ? bk  : bv);

  const int och0 = ot * 16;
  const int p0   = pt * 16;
  const int b    = p0 >> 12;          // pixel tiles never straddle a batch
  const int n    = p0 & (NPIX - 1);

  const _Float16* wrow = W + (size_t)(och0 + lm) * CDIM;            // A row
  const _Float16* xrow = X + ((size_t)b * NPIX + n + lm) * CDIM;    // B col (pixel)

  v8f acc = {};
#pragma unroll
  for (int kk = 0; kk < 8; ++kk) {
    const int c0 = kk * 32;
    // A (16-bit 16x32): lane=M, element i -> K = hi*16 + i  (contiguous, matches B)
    v8h alo = *(const v8h*)(wrow + c0 + hi * 16);
    v8h ahi = *(const v8h*)(wrow + c0 + hi * 16 + 8);
    // B (16-bit 32x16): lane=N, element i -> K = hi*16 + i  (contiguous!)
    v16h bx = *(const v16h*)(xrow + c0 + hi * 16);
    acc = wmma_f16(cat8(alo, ahi), bx, acc);
  }

#pragma unroll
  for (int r = 0; r < 8; ++r) {
    const int och = och0 + r + 8 * hi;            // C/D: m = r + 8*hi, col = lm
    const float y = acc[r] + bias[och];
    const int nn  = n + lm;
    if (which == 0) {
      const int h = och >> 4, kd = och & 15;
      Qh[(((size_t)b * NHEADS + h) * NPIX + nn) * KD + kd] = (_Float16)y;
    } else if (which == 1) {
      const int h = och >> 4, kd = och & 15;
      Kh[(((size_t)b * NHEADS + h) * NPIX + nn) * KD + kd] = (_Float16)y;
    } else {
      const int h = och >> 5, d = och & 31;
      Vt[(((size_t)b * NHEADS + h) * DV + d) * NPIX + nn] = (_Float16)y;   // [bh][d][n]
    }
  }
}

// ---------------------------------------------------------------------------
// Fused flash attention + ReLU.
// Block = 8 waves = 128 Q rows of one (b,h). All waves share the key stream,
// so K/V 32-key tiles are staged into LDS with double-buffered
// global_load_async_to_lds_b128 (ASYNCcnt) overlapping the WMMA work.
// Per wave: 2 WMMAs for S (kd=16 zero-padded to depth 32), online f32
// softmax with half-wave shuffle reductions, P transposed C-frag -> A-frag
// through private LDS, 2 WMMAs for P*V.
// Output Xt: [b][n][c] f16 with c = h*32 + d, ReLU applied.
// ---------------------------------------------------------------------------
__global__ void k_attn(const _Float16* __restrict__ Qh, const _Float16* __restrict__ Kh,
                       const _Float16* __restrict__ Vt, _Float16* __restrict__ Xt) {
  __shared__ __align__(64) _Float16 kbuf[2][32][16];   // [buf][key][kd]   2 KB
  __shared__ __align__(64) _Float16 vbuf[2][32][32];   // [buf][d][key]    4 KB
  __shared__ __align__(64) _Float16 plds[8][16][32];   // per-wave P tile  8 KB

  const int tid  = threadIdx.x;
  const int wv   = tid >> 5;
  const int lane = tid & 31;
  const int hi   = lane >> 4;
  const int lm   = lane & 15;

  const int bh     = blockIdx.x >> 5;           // b*8 + h
  const int rowblk = blockIdx.x & 31;
  const int n0     = rowblk * 128 + wv * 16;
  const int b      = bh >> 3;
  const int h      = bh & 7;

  const _Float16* Kbase = Kh + (size_t)bh * NPIX * KD;   // tile j0: +j0*KD, 1KB contig
  const _Float16* Vbase = Vt + (size_t)bh * DV * NPIX;   // row d: +d*NPIX

  // Cooperative async stage of one 32-key tile (K: 1KB, V: 2KB) into buffer bi.
  // 192 lanes x 16B. Generic->LDS: low 32 bits of a __shared__ pointer are the
  // LDS byte address (ISA 10.2 aperture rule).
  auto stage = [&](int j0, int bi) {
    if (tid < 64) {
      unsigned lds = (unsigned)(unsigned long long)&kbuf[bi][0][0] + tid * 16;
      unsigned long long ga = (unsigned long long)(Kbase + (size_t)j0 * KD) + tid * 16;
      asm volatile("global_load_async_to_lds_b128 %0, %1, off"
                   :: "v"(lds), "v"(ga) : "memory");
    } else if (tid < 192) {
      const int u = tid - 64;                 // 0..127
      const int d = u >> 2, ch = u & 3;       // 32 d-rows x 64B
      unsigned lds = (unsigned)(unsigned long long)&vbuf[bi][d][0] + ch * 16;
      unsigned long long ga =
          (unsigned long long)(Vbase + (size_t)d * NPIX + j0) + ch * 16;
      asm volatile("global_load_async_to_lds_b128 %0, %1, off"
                   :: "v"(lds), "v"(ga) : "memory");
    }
  };

  // --- Q A-fragment: rows n0..n0+15, depth kd=16 zero-padded to 32 ---------
  v16h aq;
  {
    const _Float16* qb = Qh + ((size_t)bh * NPIX + n0 + lm) * KD;
    v8h qlo = *(const v8h*)(qb);              // K = 0..7 (half 0)
    v8h qhi = *(const v8h*)(qb + 8);          // K = 8..15 (half 0)
    v8h qz  = {};                             // K = 16..31 zero pad (half 1)
    aq = hi ? cat8(qz, qz) : cat8(qlo, qhi);
  }

  v8f o0 = {}, o1 = {};
  float mr[8], lr[8];
#pragma unroll
  for (int r = 0; r < 8; ++r) { mr[r] = -3.0e38f; lr[r] = 0.0f; }

  stage(0, 0);
  asm volatile("s_wait_asynccnt 0" ::: "memory");
  __syncthreads();

  for (int it = 0; it < NPIX / 32; ++it) {
    const int bi = it & 1;
    if (it + 1 < NPIX / 32) stage((it + 1) * 32, bi ^ 1);   // prefetch next tile

    // --- K^T B-fragments (depth 32: kd 0..15 real, 16..31 zero pad) --------
    v16h bk0 = {}, bk1 = {};
    if (hi == 0) {                           // lanes 16-31 carry the zero pad
      bk0 = *(const v16h*)&kbuf[bi][lm][0];        // keys j0..j0+15
      bk1 = *(const v16h*)&kbuf[bi][16 + lm][0];   // keys j0+16..j0+31
    }
    const v8f cz = {};
    v8f s0 = wmma_f16(aq, bk0, cz);
    v8f s1 = wmma_f16(aq, bk1, cz);

    // --- online softmax (row = r + 8*hi, cols spread over 16 lanes) --------
    float tm[8];
#pragma unroll
    for (int r = 0; r < 8; ++r) tm[r] = fmaxf(s0[r], s1[r]);
#pragma unroll
    for (int m = 1; m < 16; m <<= 1) {
#pragma unroll
      for (int r = 0; r < 8; ++r) tm[r] = fmaxf(tm[r], __shfl_xor(tm[r], m, 32));
    }
    float p0[8], p1[8];
#pragma unroll
    for (int r = 0; r < 8; ++r) {
      const float mo = mr[r];
      const float mn = fmaxf(mo, tm[r]);
      mr[r] = mn;
      const float sc = __expf(mo - mn);      // uniform across the 16-lane half
      p0[r] = __expf(s0[r] - mn);
      p1[r] = __expf(s1[r] - mn);
      lr[r] = lr[r] * sc + p0[r] + p1[r];    // lane-distributed row sum
      o0[r] *= sc;  o1[r] *= sc;
    }

    // --- transpose P: C-fragment -> A-fragment via private LDS -------------
#pragma unroll
    for (int r = 0; r < 8; ++r) {
      plds[wv][r + 8 * hi][lm]      = (_Float16)p0[r];
      plds[wv][r + 8 * hi][lm + 16] = (_Float16)p1[r];
    }
    asm volatile("s_wait_dscnt 0" ::: "memory");
    v8h pl = *(const v8h*)&plds[wv][lm][hi * 16];       // K = hi*16+..
    v8h ph = *(const v8h*)&plds[wv][lm][hi * 16 + 8];   // K = hi*16+8+..
    v16h pa = cat8(pl, ph);

    // --- V B-fragments from LDS: [d][key] so per-lane contiguous -----------
    v16h bv0 = *(const v16h*)&vbuf[bi][lm][hi * 16];        // d = lm
    v16h bv1 = *(const v16h*)&vbuf[bi][lm + 16][hi * 16];   // d = lm+16

    o0 = wmma_f16(pa, bv0, o0);
    o1 = wmma_f16(pa, bv1, o1);

    asm volatile("s_wait_asynccnt 0" ::: "memory");  // own DMA for tile it+1 done
    __syncthreads();                                 // publish buffer, release old
  }

  // --- finalize: reduce l, normalize, ReLU, store [b][n][c=h*32+d] ---------
#pragma unroll
  for (int m = 1; m < 16; m <<= 1) {
#pragma unroll
    for (int r = 0; r < 8; ++r) lr[r] += __shfl_xor(lr[r], m, 32);
  }
#pragma unroll
  for (int r = 0; r < 8; ++r) {
    const float inv = 1.0f / lr[r];
    float y0 = o0[r] * inv;  y0 = (y0 > 0.0f) ? y0 : 0.0f;
    float y1 = o1[r] * inv;  y1 = (y1 > 0.0f) ? y1 : 0.0f;
    const int nrow = n0 + r + 8 * hi;
    _Float16* xrow = Xt + ((size_t)b * NPIX + nrow) * CDIM + h * DV;
    xrow[lm]      = (_Float16)y0;
    xrow[lm + 16] = (_Float16)y1;
  }
}

// ---------------------------------------------------------------------------
// Output projection via WMMA. Xt is pixel-major so B-fragments are contiguous.
// out f32 [B][256][4096].
// ---------------------------------------------------------------------------
__global__ void k_proj(const _Float16* __restrict__ Xt, const _Float16* __restrict__ Wp2,
                       const float* __restrict__ bp, float* __restrict__ out) {
  const int wv   = threadIdx.x >> 5;
  const int lane = threadIdx.x & 31;
  const int hi   = lane >> 4;
  const int lm   = lane & 15;
  const int t    = blockIdx.x * 8 + wv;    // 0..8191
  const int ot   = t >> 9;                 // out-channel tile, 0..15
  const int pt   = t & 511;                // pixel tile
  const int och0 = ot * 16;
  const int p0   = pt * 16;
  const int b    = p0 >> 12;
  const int n    = p0 & (NPIX - 1);

  const _Float16* wrow = Wp2 + (size_t)(och0 + lm) * CDIM;
  const _Float16* xrow = Xt + ((size_t)b * NPIX + n + lm) * CDIM;

  v8f acc = {};
#pragma unroll
  for (int kk = 0; kk < 8; ++kk) {
    const int c0 = kk * 32;
    v8h alo = *(const v8h*)(wrow + c0 + hi * 16);
    v8h ahi = *(const v8h*)(wrow + c0 + hi * 16 + 8);
    v16h bx = *(const v16h*)(xrow + c0 + hi * 16);
    acc = wmma_f16(cat8(alo, ahi), bx, acc);
  }
#pragma unroll
  for (int r = 0; r < 8; ++r) {
    const int och = och0 + r + 8 * hi;
    out[((size_t)b * CDIM + och) * NPIX + n + lm] = acc[r] + bp[och];
  }
}

// ---------------------------------------------------------------------------
extern "C" void kernel_launch(void* const* d_in, const int* in_sizes, int n_in,
                              void* d_out, int out_size, void* d_ws, size_t ws_size,
                              hipStream_t stream) {
  (void)in_sizes; (void)n_in; (void)out_size; (void)ws_size;

  const float* rgb  = (const float*)d_in[0];
  const float* edge = (const float*)d_in[1];
  const float* Wq   = (const float*)d_in[2];
  const float* sq   = (const float*)d_in[3];
  const float* bq   = (const float*)d_in[4];
  const float* Wk   = (const float*)d_in[5];
  const float* sk   = (const float*)d_in[6];
  const float* bk   = (const float*)d_in[7];
  const float* Wv   = (const float*)d_in[8];
  const float* sv   = (const float*)d_in[9];
  const float* bv   = (const float*)d_in[10];
  const float* Wp   = (const float*)d_in[11];
  const float* sp   = (const float*)d_in[12];
  const float* bp   = (const float*)d_in[13];
  float* out = (float*)d_out;

  char* ws = (char*)d_ws;
  _Float16* xrT = (_Float16*)(ws + 0);                 // [b][n][c] f16, 4 MB
  _Float16* xeT = (_Float16*)(ws + 4194304);           // 4 MB
  _Float16* Wq2 = (_Float16*)(ws + 8388608);           // 64 KB
  _Float16* Wk2 = (_Float16*)(ws + 8454144);           // 64 KB
  _Float16* Wv2 = (_Float16*)(ws + 8519680);           // 128 KB
  _Float16* Wp2 = (_Float16*)(ws + 8650752);           // 128 KB
  _Float16* Qh  = (_Float16*)(ws + 8781824);           // [bh][n][kd], 2 MB
  _Float16* Kh  = (_Float16*)(ws + 10878976);          // [bh][n][kd], 2 MB
  _Float16* Vt  = (_Float16*)(ws + 12976128);          // [bh][d][n],  4 MB
  _Float16* Xt  = (_Float16*)(ws + 17170432);          // [b][n][c],   4 MB (end ~21.4 MB)

  const int nin = BATCH * NPIX * CDIM;                 // 2,097,152

  k_fold_w<<<(128 * 256 + 255) / 256, 256, 0, stream>>>(Wq, sq, Wq2, 128, 256);
  k_fold_w<<<(128 * 256 + 255) / 256, 256, 0, stream>>>(Wk, sk, Wk2, 128, 256);
  k_fold_w<<<(256 * 256 + 255) / 256, 256, 0, stream>>>(Wv, sv, Wv2, 256, 256);
  k_fold_w<<<(256 * 256 + 255) / 256, 256, 0, stream>>>(Wp, sp, Wp2, 256, 256);
  k_cvt_T<<<(nin + 255) / 256, 256, 0, stream>>>(rgb,  xrT);
  k_cvt_T<<<(nin + 255) / 256, 256, 0, stream>>>(edge, xeT);

  // (8 Q + 8 K + 16 V out-ch tiles) * 512 pixel tiles = 16384 waves
  k_qkv<<<2048, 256, 0, stream>>>(xrT, xeT, Wq2, Wk2, Wv2, bq, bk, bv, Qh, Kh, Vt);

  // 16 (b,h) * 32 row-blocks = 512 blocks, 8 waves (128 Q rows) each
  k_attn<<<512, 256, 0, stream>>>(Qh, Kh, Vt, Xt);

  // 16 out-ch tiles * 512 pixel tiles = 8192 waves
  k_proj<<<1024, 256, 0, stream>>>(Xt, Wp2, bp, out);
}
